// c2vRNNModel_22445499089358
// MI455X (gfx1250) — matrix operand order repaired and checked by reading, not compiled, not run on hardware
//
#include <hip/hip_runtime.h>
#include <hip/hip_bf16.h>
#include <math.h>

// ---------------- model constants ----------------
#define BATCH   16
#define SEQ     100
#define CLEN    100
#define IN_DIM  20
#define EMBED   100
#define HID     128
#define OUTD    10
#define DDIM    320           // IN_DIM + 3*EMBED
#define RNN_IN  340           // 2*IN_DIM + 3*EMBED
#define RNN_KP  352           // RNN_IN padded to 32
#define NROWS   (BATCH*SEQ*CLEN)   // 160000 GEMM rows
#define GATES   (4*HID)            // 512

typedef __attribute__((ext_vector_type(16))) __bf16        v16bf;
typedef __attribute__((ext_vector_type(8)))  float         v8f;
typedef __attribute__((ext_vector_type(8)))  unsigned int  v8u;

__device__ __forceinline__ float sigf(float x) { return 1.0f / (1.0f + __expf(-x)); }

// Branch-free tanh: native v_tanh_f32 on gfx1250 (confirmed in round-2 disasm).
__device__ __forceinline__ float tanh_fast(float x) {
#if __has_builtin(__builtin_amdgcn_tanhf)
  return __builtin_amdgcn_tanhf(x);
#else
  float e = __expf(2.0f * x);
  return 1.0f - 2.0f / (e + 1.0f);
#endif
}

// A-fragment (16-bit A 16x32, ISA 7.12.2): lane holds 8 contiguous bf16 at kbase,
// plus 8 contiguous bf16 at kbase+16.  p points at the lane's kbase element.
__device__ __forceinline__ v16bf load_a_frag(const __bf16* p) {
  uint4 lo = *(const uint4*)(p);
  uint4 hi = *(const uint4*)(p + 16);
  v8u u = { lo.x, lo.y, lo.z, lo.w, hi.x, hi.y, hi.z, hi.w };
  return __builtin_bit_cast(v16bf, u);
}

// B-fragment from pre-packed buffer: 16 contiguous bf16 per lane.
__device__ __forceinline__ v16bf load_b_frag(const __bf16* p) {
  uint4 lo = *(const uint4*)(p);
  uint4 hi = *(const uint4*)(p + 8);
  v8u u = { lo.x, lo.y, lo.z, lo.w, hi.x, hi.y, hi.z, hi.w };
  return __builtin_bit_cast(v16bf, u);
}

// ---------------------------------------------------------------------------
// Pack a row-major f32 weight W[N][K] into fragment-ready bf16 B-tiles:
// dst[((kt*NT + nt)*32 + lane)*16 + i] = W[nt*16 + (lane&15)][kt*32 + (lane>=16?8:0) + (i&7) + (i>=8?16:0)]
// (16-bit B 32x16 layout: lanes 0-15 hold K=0..7,16..23; lanes 16-31 K=8..15,24..31)
// ---------------------------------------------------------------------------
__global__ void pack_b_kernel(const float* __restrict__ src, __bf16* __restrict__ dst,
                              int N, int K, int KT) {
  int NT = N >> 4;
  int total = KT * NT * 32 * 16;
  int tid = blockIdx.x * blockDim.x + threadIdx.x;
  if (tid >= total) return;
  int i    = tid & 15;
  int lane = (tid >> 4) & 31;
  int f    = tid >> 9;
  int nt   = f % NT;
  int kt   = f / NT;
  int k = kt * 32 + ((lane >= 16) ? 8 : 0) + (i & 7) + ((i >= 8) ? 16 : 0);
  int n = nt * 16 + (lane & 15);
  float v = (k < K) ? src[(size_t)n * K + k] : 0.0f;
  dst[tid] = (__bf16)v;
}

// ---------------------------------------------------------------------------
// Build full[row][0:320] = [sn(100) | en(100) | pe(100) | first(20)] in bf16.
// One wave per row (= one (b,s,ctx) triple). Embedding tables are L2-resident.
// ---------------------------------------------------------------------------
__global__ void __launch_bounds__(256) build_full_kernel(
    const float* __restrict__ x, const float* __restrict__ emb_n,
    const float* __restrict__ emb_p, __bf16* __restrict__ fullb) {
  int row  = blockIdx.x * 8 + (threadIdx.x >> 5);
  int lane = threadIdx.x & 31;
  int bs   = row / CLEN;
  int ctx  = row - bs * CLEN;
  const float* xr = x + (size_t)bs * DDIM;
  int i0 = (int)xr[IN_DIM + ctx * 3 + 0];   // start  -> embed_nodes
  int i1 = (int)xr[IN_DIM + ctx * 3 + 1];   // path   -> embed_paths
  int i2 = (int)xr[IN_DIM + ctx * 3 + 2];   // end    -> embed_nodes
  __bf16* dst = fullb + (size_t)row * DDIM;
  for (int i = lane; i < DDIM; i += 32) {
    float v;
    if      (i < 100) v = emb_n[(size_t)i0 * EMBED + i];
    else if (i < 200) v = emb_n[(size_t)i2 * EMBED + (i - 100)];
    else if (i < 300) v = emb_p[(size_t)i1 * EMBED + (i - 200)];
    else              v = xr[i - 300];
    dst[i] = (__bf16)v;
  }
}

// ---------------------------------------------------------------------------
// cw[m] = sum_j W_attn[j] * tanh( (full @ W_trans^T)[m,j] + b_trans[j] ) + b_attn
// WMMA bf16 GEMM, M-tile per wave (16 rows), N=320 (20 tiles), K=320 (10 steps).
// B-tiles staged in LDS (double buffered, shared by 8 waves); B-fragments are
// additionally software-pipelined one step ahead in registers so ds_read
// overlaps the previous WMMA instead of draining DScnt before each one.
// tanh + attention dot fused into the accumulator epilogue; t is never stored.
// ---------------------------------------------------------------------------
__global__ void __launch_bounds__(256) attn_gemm_kernel(
    const __bf16* __restrict__ fullb, const __bf16* __restrict__ pWt,
    const float* __restrict__ b_trans, const float* __restrict__ W_attn,
    const float* __restrict__ b_attn, float* __restrict__ cw) {
  __shared__ __bf16 Bs[2][10 * 32 * 16];    // double-buffered nt tile, 2 x 10 KB

  int tid  = threadIdx.x;
  int wave = blockIdx.x * 8 + (tid >> 5);
  int lane = tid & 31;
  size_t row0 = (size_t)wave * 16;
  int arow = lane & 15;
  int kb   = (lane >> 4) * 8;

  // Keep all 10 A k-fragments (this wave's 16x320 slab) resident.
  const __bf16* abase = fullb + (row0 + arow) * DDIM;
  v16bf A[10];
#pragma unroll
  for (int kk = 0; kk < 10; ++kk) A[kk] = load_a_frag(abase + kk * 32 + kb);

  // Cooperative stage of one nt-tile (10 kk-frags x 32 lanes x 32 B) into LDS.
  auto stage = [&](int nt, int buf) {
    for (int c = tid; c < 10 * 32; c += 256) {
      int kk = c >> 5, ln = c & 31;
      const uint4* src = (const uint4*)(pWt + ((size_t)(kk * 20 + nt) * 32 + ln) * 16);
      uint4* dst = (uint4*)(&Bs[buf][(kk * 32 + ln) * 16]);
      dst[0] = src[0];
      dst[1] = src[1];
    }
  };

  stage(0, 0);
  float rowacc[8] = {0.f, 0.f, 0.f, 0.f, 0.f, 0.f, 0.f, 0.f};

  for (int nt = 0; nt < 20; ++nt) {
    int buf = nt & 1;
    __syncthreads();                         // staged buf is ready
    if (nt + 1 < 20) stage(nt + 1, buf ^ 1); // prefetch next tile into other buf

    const __bf16* bs = &Bs[buf][lane * 16];
    v16bf Bcur = load_b_frag(bs);
    v8f acc = {};
#pragma unroll
    for (int kk = 0; kk < 10; ++kk) {
      v16bf Bnext = Bcur;
      if (kk < 9) Bnext = load_b_frag(bs + (size_t)(kk + 1) * 32 * 16);
      acc = __builtin_amdgcn_wmma_f32_16x16x32_bf16(false, A[kk], false, Bcur,
                                                    (short)0, acc, false, false);
      Bcur = Bnext;
    }

    int n = nt * 16 + arow;          // accumulator column
    float bt = b_trans[n];
    float wa = W_attn[n];
#pragma unroll
    for (int r = 0; r < 8; ++r)
      rowacc[r] += tanh_fast(acc[r] + bt) * wa;
  }

  // Reduce each row across the 16 lanes of each half (C layout: lane half = row +8).
  float ba = b_attn[0];
#pragma unroll
  for (int r = 0; r < 8; ++r) {
    float v = rowacc[r];
    v += __shfl_xor(v, 1);
    v += __shfl_xor(v, 2);
    v += __shfl_xor(v, 4);
    v += __shfl_xor(v, 8);
    if (arow == 0)
      cw[row0 + (size_t)((lane >> 4) * 8 + r)] = v + ba;
  }
}

// ---------------------------------------------------------------------------
// Softmax over axis=1 (the S axis!) per (b, ctx): 1600 groups of 100, stride CLEN.
// In-place cw -> aw.
// ---------------------------------------------------------------------------
__global__ void __launch_bounds__(128) softmax_s_kernel(float* __restrict__ cw) {
  int b = blockIdx.x / CLEN, ctx = blockIdx.x % CLEN;
  __shared__ float red[128];
  int t = threadIdx.x;
  size_t base = (size_t)b * SEQ * CLEN + ctx;
  float v = -3.0e38f;
  if (t < SEQ) v = cw[base + (size_t)t * CLEN];
  red[t] = v;
  __syncthreads();
  for (int s2 = 64; s2 > 0; s2 >>= 1) {
    if (t < s2) red[t] = fmaxf(red[t], red[t + s2]);
    __syncthreads();
  }
  float mx = red[0];
  __syncthreads();
  float e = (t < SEQ) ? __expf(v - mx) : 0.0f;
  red[t] = e;
  __syncthreads();
  for (int s2 = 64; s2 > 0; s2 >>= 1) {
    if (t < s2) red[t] += red[t + s2];
    __syncthreads();
  }
  float inv = 1.0f / red[0];
  if (t < SEQ) cw[base + (size_t)t * CLEN] = e * inv;
}

// ---------------------------------------------------------------------------
// code_vec[b,s,d] = sum_ctx full[b,s,ctx,d] * aw[b,s,ctx]; emit rnn_in = [first | code_vec].
// One block per (b,s), 320 threads (one per d).
// ---------------------------------------------------------------------------
__global__ void __launch_bounds__(320) code_vec_kernel(
    const __bf16* __restrict__ fullb, const float* __restrict__ aw,
    const float* __restrict__ x, float* __restrict__ rnn_in) {
  int row = blockIdx.x;           // b*SEQ + s
  int d   = threadIdx.x;
  __shared__ float sa[CLEN];
  if (d < CLEN) sa[d] = aw[(size_t)row * CLEN + d];
  __syncthreads();
  const __bf16* fb = fullb + (size_t)row * CLEN * DDIM + d;
  float sum = 0.0f;
#pragma unroll 4
  for (int c = 0; c < CLEN; ++c) sum += (float)fb[(size_t)c * DDIM] * sa[c];
  rnn_in[(size_t)row * RNN_IN + IN_DIM + d] = sum;
  if (d < IN_DIM) rnn_in[(size_t)row * RNN_IN + d] = x[(size_t)row * DDIM + d];
}

// ---------------------------------------------------------------------------
// Persistent single-workgroup LSTM over 100 steps + fused final FC/sigmoid.
// Gates GEMM (M=16 batch, N=512 gates, K=352+128) via WMMA bf16; 8 waves x 4 N-tiles.
// A (xt/h) from LDS, B (weights) from L2-hot global, both pipelined one step.
// h/c/gates staged in LDS; writes d_out directly.
// ---------------------------------------------------------------------------
__global__ void __launch_bounds__(256) lstm_kernel(
    const float* __restrict__ rnn_in, const __bf16* __restrict__ pWih,
    const __bf16* __restrict__ pWhh, const float* __restrict__ b_ih,
    const float* __restrict__ b_hh, const float* __restrict__ W_fc,
    const float* __restrict__ b_fc, float* __restrict__ out) {
  __shared__ __bf16 xt[BATCH * RNN_KP];     // 16x352 bf16 (padded input row)
  __shared__ __bf16 hb[BATCH * HID];        // h in bf16 for WMMA
  __shared__ float  hf[BATCH * HID];        // h in f32
  __shared__ float  cf[BATCH * HID];        // c in f32
  __shared__ float  g [BATCH * GATES];      // 16x512 gate pre-activations

  int tid  = threadIdx.x;
  int lane = tid & 31;
  int wave = tid >> 5;
  int arow = lane & 15;
  int kb   = (lane >> 4) * 8;
  int mb   = (lane >> 4) * 8;               // accumulator row base for this lane half

  for (int i = tid; i < BATCH * HID; i += 256) {
    hf[i] = 0.0f; cf[i] = 0.0f; hb[i] = (__bf16)0.0f;
  }
  __syncthreads();

  for (int s = 0; s < SEQ; ++s) {
    // Stage this step's 16x340 input rows into LDS (bf16, zero-padded to 352).
    for (int i = tid; i < BATCH * RNN_KP; i += 256) {
      int b = i / RNN_KP, k = i - b * RNN_KP;
      float v = (k < RNN_IN) ? rnn_in[((size_t)b * SEQ + s) * RNN_IN + k] : 0.0f;
      xt[i] = (__bf16)v;
    }
    __syncthreads();

    // Gates GEMM: each wave computes 4 N-tiles of 16x16 (K pipelined one step).
    for (int nn = 0; nn < 4; ++nn) {
      int nt = wave * 4 + nn;
      v8f acc = {};
      v16bf Acur = load_a_frag(&xt[arow * RNN_KP + kb]);
      v16bf Bcur = load_b_frag(&pWih[((size_t)nt * 32 + lane) * 16]);
#pragma unroll
      for (int kk = 0; kk < RNN_KP / 32; ++kk) {            // 11 k-steps, input part
        v16bf Anext = Acur, Bnext = Bcur;
        if (kk + 1 < RNN_KP / 32) {
          Anext = load_a_frag(&xt[arow * RNN_KP + (kk + 1) * 32 + kb]);
          Bnext = load_b_frag(&pWih[((size_t)((kk + 1) * 32 + nt) * 32 + lane) * 16]);
        } else {
          Anext = load_a_frag(&hb[arow * HID + kb]);
          Bnext = load_b_frag(&pWhh[((size_t)nt * 32 + lane) * 16]);
        }
        acc = __builtin_amdgcn_wmma_f32_16x16x32_bf16(false, Acur, false, Bcur,
                                                      (short)0, acc, false, false);
        Acur = Anext; Bcur = Bnext;
      }
#pragma unroll
      for (int kk = 0; kk < HID / 32; ++kk) {               // 4 k-steps, hidden part
        v16bf Anext = Acur, Bnext = Bcur;
        if (kk + 1 < HID / 32) {
          Anext = load_a_frag(&hb[arow * HID + (kk + 1) * 32 + kb]);
          Bnext = load_b_frag(&pWhh[((size_t)((kk + 1) * 32 + nt) * 32 + lane) * 16]);
        }
        acc = __builtin_amdgcn_wmma_f32_16x16x32_bf16(false, Acur, false, Bcur,
                                                      (short)0, acc, false, false);
        Acur = Anext; Bcur = Bnext;
      }
      int n = nt * 16 + arow;
      float bias = b_ih[n] + b_hh[n];
#pragma unroll
      for (int r = 0; r < 8; ++r) g[(mb + r) * GATES + n] = acc[r] + bias;
    }
    __syncthreads();

    // Elementwise LSTM cell update (gate order i, f, g, o).
    for (int i = tid; i < BATCH * HID; i += 256) {
      int b = i >> 7, j = i & (HID - 1);
      float gi = g[b * GATES + j];
      float gf = g[b * GATES + HID + j];
      float gg = g[b * GATES + 2 * HID + j];
      float go = g[b * GATES + 3 * HID + j];
      float c  = sigf(gf) * cf[i] + sigf(gi) * tanh_fast(gg);
      float h  = sigf(go) * tanh_fast(c);
      cf[i] = c; hf[i] = h; hb[i] = (__bf16)h;
    }
    __syncthreads();

    // Fused FC (128 -> 10) + sigmoid, straight to d_out.
    if (tid < BATCH * OUTD) {
      int b = tid / OUTD, o = tid % OUTD;
      float acc = b_fc[o];
#pragma unroll 8
      for (int j = 0; j < HID; ++j) acc += hf[b * HID + j] * W_fc[o * HID + j];
      out[((size_t)b * SEQ + s) * OUTD + o] = sigf(acc);
    }
    __syncthreads();
  }
}

// ---------------------------------------------------------------------------
extern "C" void kernel_launch(void* const* d_in, const int* in_sizes, int n_in,
                              void* d_out, int out_size, void* d_ws, size_t ws_size,
                              hipStream_t stream) {
  (void)in_sizes; (void)n_in; (void)out_size; (void)ws_size;
  const float* x       = (const float*)d_in[0];
  const float* emb_n   = (const float*)d_in[1];
  const float* emb_p   = (const float*)d_in[2];
  const float* W_trans = (const float*)d_in[3];
  const float* b_trans = (const float*)d_in[4];
  const float* W_attn  = (const float*)d_in[5];
  const float* b_attn  = (const float*)d_in[6];
  const float* W_ih    = (const float*)d_in[7];
  const float* W_hh    = (const float*)d_in[8];
  const float* b_ih    = (const float*)d_in[9];
  const float* b_hh    = (const float*)d_in[10];
  const float* W_fc    = (const float*)d_in[11];
  const float* b_fc    = (const float*)d_in[12];
  float* out = (float*)d_out;

  char* ws = (char*)d_ws;
  size_t off = 0;
  auto take = [&](size_t nbytes) -> void* {
    void* p = ws + off;
    off = (off + nbytes + 255) & ~(size_t)255;
    return p;
  };
  __bf16* fullb = (__bf16*)take((size_t)NROWS * DDIM * 2);     // ~102.4 MB
  __bf16* pWt   = (__bf16*)take((size_t)10 * 20 * 32 * 16 * 2);
  __bf16* pWih  = (__bf16*)take((size_t)11 * 32 * 32 * 16 * 2);
  __bf16* pWhh  = (__bf16*)take((size_t)4  * 32 * 32 * 16 * 2);
  float*  cw    = (float*)take((size_t)NROWS * 4);
  float*  rnnin = (float*)take((size_t)BATCH * SEQ * RNN_IN * 4);

  // Pack weights into WMMA B-fragment order (bf16).
  pack_b_kernel<<<(10 * 20 * 512 + 255) / 256, 256, 0, stream>>>(W_trans, pWt, DDIM, DDIM, 10);
  pack_b_kernel<<<(11 * 32 * 512 + 255) / 256, 256, 0, stream>>>(W_ih, pWih, GATES, RNN_IN, 11);
  pack_b_kernel<<<(4  * 32 * 512 + 255) / 256, 256, 0, stream>>>(W_hh, pWhh, GATES, HID, 4);

  // Gather embeddings -> full (bf16).
  build_full_kernel<<<NROWS / 8, 256, 0, stream>>>(x, emb_n, emb_p, fullb);

  // Fused GEMM + tanh + attention-logit reduction (LDS-staged B, double buffered).
  attn_gemm_kernel<<<(NROWS / 16) / 8, 256, 0, stream>>>(fullb, pWt, b_trans, W_attn, b_attn, cw);

  // Softmax over the S axis (matches reference axis=1), in place.
  softmax_s_kernel<<<BATCH * CLEN, 128, 0, stream>>>(cw);

  // Attention-weighted context vector + rnn input assembly.
  code_vec_kernel<<<BATCH * SEQ, DDIM, 0, stream>>>(fullb, cw, x, rnnin);

  // Sequential LSTM + fused output FC.
  lstm_kernel<<<1, 256, 0, stream>>>(rnnin, pWih, pWhh, b_ih, b_hh, W_fc, b_fc, out);
}
